// RNNDecoder_70652212019654
// MI455X (gfx1250) — compile-verified, matrix operand order checked by reading, not run on hardware
//
#include <hip/hip_runtime.h>
#include <hip/hip_bf16.h>

// ---------------------------------------------------------------------------
// Problem constants (fixed by the reference): B=32, S=64, H=E=512, L=2, V=32000
// ---------------------------------------------------------------------------
#define BB 32
#define SS 64
#define HH 512
#define EE 512
#define VV 32000

typedef __attribute__((ext_vector_type(16))) _Float16 v16h;
typedef __attribute__((ext_vector_type(8)))  float    v8f;

// ---------------------------------------------------------------------------
// Swizzle an (R x K) fp32 matrix (row-major) into packed f16 WMMA operand
// tiles. Tile (rt,kt) covers rows rt*16..+15 and cols kt*32..+31 (K dim).
// Per-lane element e -> K = Kmap(lane,e) = (e<8 ? e : e+8) + ((lane>=16)?8:0)
// (16-bit operand layout per CDNA5 ISA 7.12.2), row = rt*16 + lane%16.
// Each lane's 16 f16 values are contiguous: one 32B v16h load per tile in the
// GEMM. Used for both weights (R = GEMM N dim, once per launch) and
// activations (R = M = 32, once per GEMM).
// ---------------------------------------------------------------------------
__global__ __launch_bounds__(32)
void k_swizzle(const float* __restrict__ W, _Float16* __restrict__ out, int K)
{
    const int kt   = blockIdx.x;
    const int rt   = blockIdx.y;
    const int lane = threadIdx.x;
    const int ktiles = K >> 5;
    const int r     = rt * 16 + (lane & 15);
    const int kbase = kt * 32 + ((lane >> 4) << 3);
    _Float16* o = out + ((size_t)rt * ktiles + kt) * 512 + (size_t)lane * 16;
    const float* w = W + (size_t)r * K + kbase;
#pragma unroll
    for (int e = 0; e < 8; ++e) o[e]     = (_Float16)w[e];
#pragma unroll
    for (int e = 0; e < 8; ++e) o[8 + e] = (_Float16)w[16 + e];
}

// ---------------------------------------------------------------------------
// WMMA GEMM: C(M=32, N) = A @ W^T + bias, both operands pre-swizzled f16.
// 4 waves / block; each wave computes a full 32x32 C macro-tile (2 M-tiles x
// 2 N-tiles) with FOUR independent accumulator chains. Each weight (B) tile
// is loaded by exactly ONE wave -> weight traffic (the M=32 roofline bound,
// 32.8 MB of W_out per step from L2) is paid once, and 4 chains hide WMMA
// latency. Inner loop: four 32B loads + four v_wmma_f32_16x16x32_f16.
// ---------------------------------------------------------------------------
__global__ __launch_bounds__(128)
void k_wmma_gemm(const _Float16* __restrict__ Asw,
                 const _Float16* __restrict__ Bsw,
                 const float* __restrict__ bias,
                 float* __restrict__ C, int N, int K)
{
    const int lane = threadIdx.x & 31;
    const int wave = threadIdx.x >> 5;
    const int nt0  = (blockIdx.x * 4 + wave) * 2;  // first of 2 N tiles
    const int ktiles = K >> 5;

    const _Float16* ap0 = Asw + (size_t)lane * 16;                 // M rows 0..15
    const _Float16* ap1 = ap0 + (size_t)ktiles * 512;              // M rows 16..31
    const _Float16* bp0 = Bsw + ((size_t)nt0 * ktiles) * 512 + (size_t)lane * 16;
    const _Float16* bp1 = bp0 + (size_t)ktiles * 512;

    v8f acc00 = {}, acc01 = {}, acc10 = {}, acc11 = {};
    for (int kt = 0; kt < ktiles; ++kt) {
        const v16h a0 = *(const v16h*)ap0;
        const v16h a1 = *(const v16h*)ap1;
        const v16h b0 = *(const v16h*)bp0;
        const v16h b1 = *(const v16h*)bp1;
        acc00 = __builtin_amdgcn_wmma_f32_16x16x32_f16(
                    false, a0, false, b0, (short)0, acc00, false, false);
        acc01 = __builtin_amdgcn_wmma_f32_16x16x32_f16(
                    false, a0, false, b1, (short)0, acc01, false, false);
        acc10 = __builtin_amdgcn_wmma_f32_16x16x32_f16(
                    false, a1, false, b0, (short)0, acc10, false, false);
        acc11 = __builtin_amdgcn_wmma_f32_16x16x32_f16(
                    false, a1, false, b1, (short)0, acc11, false, false);
        ap0 += 512; ap1 += 512; bp0 += 512; bp1 += 512;
    }

    const int laneN = lane & 15;
    const int rhalf = (lane >> 4) << 3;            // C row half (ISA C layout)
    const int col   = nt0 * 16 + laneN;
    const float bv0 = bias[col];
    const float bv1 = bias[col + 16];
    float* cp0 = C + (size_t)rhalf * N + col;          // M rows 0..15 half
    float* cp1 = C + (size_t)(16 + rhalf) * N + col;   // M rows 16..31 half
#pragma unroll
    for (int r = 0; r < 8; ++r) {
        cp0[(size_t)r * N]      = acc00[r] + bv0;
        cp0[(size_t)r * N + 16] = acc01[r] + bv1;
        cp1[(size_t)r * N]      = acc10[r] + bv0;
        cp1[(size_t)r * N + 16] = acc11[r] + bv1;
    }
}

// ---------------------------------------------------------------------------
// Luong dot attention over top-layer hidden state: scores, softmax, context.
// One block per batch row. S=64, H=512. (2M MACs/step total -> VALU is fine.)
// ---------------------------------------------------------------------------
__global__ __launch_bounds__(256)
void k_attn(const float* __restrict__ h_top,   // B x H
            const float* __restrict__ enc,     // B x S x H
            float* __restrict__ ctx)           // B x H
{
    __shared__ float hvec[HH];
    __shared__ float attn[SS];
    __shared__ float invsum;
    const int b = blockIdx.x, t = threadIdx.x;

    for (int i = t; i < HH; i += 256) hvec[i] = h_top[b * HH + i];
    __syncthreads();

    if (t < SS) {
        const float* e = enc + ((size_t)b * SS + t) * HH;
        float s = 0.f;
        for (int i = 0; i < HH; ++i) s += hvec[i] * e[i];
        attn[t] = s;
    }
    __syncthreads();

    if (t == 0) {
        float mx = attn[0];
        for (int s = 1; s < SS; ++s) mx = fmaxf(mx, attn[s]);
        float sum = 0.f;
        for (int s = 0; s < SS; ++s) { float v = __expf(attn[s] - mx); attn[s] = v; sum += v; }
        invsum = 1.f / sum;
    }
    __syncthreads();

    const float inv = invsum;
    for (int hh = t; hh < HH; hh += 256) {
        const float* e = enc + (size_t)b * SS * HH + hh;
        float acc = 0.f;
        for (int s = 0; s < SS; ++s) acc += attn[s] * e[(size_t)s * HH];
        ctx[b * HH + hh] = acc * inv;
    }
}

// ---------------------------------------------------------------------------
// Embedding gather + build xcat = [emb | ctx] (B x (E+H)); also keep emb.
// ---------------------------------------------------------------------------
__global__ __launch_bounds__(256)
void k_embed_concat(const int* __restrict__ inputs, int t,
                    const float* __restrict__ emb_table,
                    const float* __restrict__ ctx,
                    float* __restrict__ xcat,   // B x 1024
                    float* __restrict__ emb)    // B x 512
{
    const int b = blockIdx.x, tid = threadIdx.x;
    const int tok = inputs[b * SS + t];
    for (int c = tid; c < EE; c += 256) {
        const float ev = emb_table[(size_t)tok * EE + c];
        emb[b * EE + c]              = ev;
        xcat[b * (EE + HH) + c]      = ev;
        xcat[b * (EE + HH) + EE + c] = ctx[b * HH + c];
    }
}

// ---------------------------------------------------------------------------
// GRU gate nonlinearity, h updated in place. gi/gh are B x 3H.
// ---------------------------------------------------------------------------
__global__ __launch_bounds__(256)
void k_gru_gates(const float* __restrict__ gi, const float* __restrict__ gh,
                 float* __restrict__ h)
{
    const int idx = blockIdx.x * 256 + threadIdx.x;   // 0 .. B*H-1
    const int b = idx >> 9, j = idx & (HH - 1);
    const float* gib = gi + (size_t)b * 3 * HH;
    const float* ghb = gh + (size_t)b * 3 * HH;
    const float ir = gib[j], iz = gib[HH + j], in = gib[2 * HH + j];
    const float hr = ghb[j], hz = ghb[HH + j], hn = ghb[2 * HH + j];
    const float r = 1.f / (1.f + __expf(-(ir + hr)));
    const float z = 1.f / (1.f + __expf(-(iz + hz)));
    const float n = tanhf(in + r * hn);
    h[idx] = (1.f - z) * n + z * h[idx];
}

// ---------------------------------------------------------------------------
// pcat = [h_top | ctx | emb]   (B x 1536)
// ---------------------------------------------------------------------------
__global__ __launch_bounds__(256)
void k_concat_proj(const float* __restrict__ h1, const float* __restrict__ ctx,
                   const float* __restrict__ emb, float* __restrict__ pcat)
{
    const int b = blockIdx.x, tid = threadIdx.x;
    for (int c = tid; c < HH; c += 256) {
        pcat[b * 1536 + c]          = h1[b * HH + c];
        pcat[b * 1536 + HH + c]     = ctx[b * HH + c];
        pcat[b * 1536 + 2 * HH + c] = emb[b * EE + c];
    }
}

// ---------------------------------------------------------------------------
// log-softmax over V=32000 + NLL(target=tok) + argmax (first-max tie-break).
// One block per batch row.
// ---------------------------------------------------------------------------
__global__ __launch_bounds__(256)
void k_lsm(const float* __restrict__ logits, const int* __restrict__ inputs,
           int t, float* __restrict__ losses, float* __restrict__ sym)
{
    __shared__ float smax[256];
    __shared__ int   simx[256];
    __shared__ float ssum[256];
    const int b = blockIdx.x, tid = threadIdx.x;
    const float* lg = logits + (size_t)b * VV;

    float mv = -__builtin_inff(); int mi = 0;
    for (int v = tid; v < VV; v += 256) {
        const float x = lg[v];
        if (x > mv) { mv = x; mi = v; }
    }
    smax[tid] = mv; simx[tid] = mi;
    __syncthreads();
    for (int off = 128; off > 0; off >>= 1) {
        if (tid < off) {
            const float o = smax[tid + off]; const int oi = simx[tid + off];
            if (o > smax[tid] || (o == smax[tid] && oi < simx[tid])) {
                smax[tid] = o; simx[tid] = oi;
            }
        }
        __syncthreads();
    }
    const float gmax = smax[0];

    float s = 0.f;
    for (int v = tid; v < VV; v += 256) s += __expf(lg[v] - gmax);
    ssum[tid] = s;
    __syncthreads();
    for (int off = 128; off > 0; off >>= 1) {
        if (tid < off) ssum[tid] += ssum[tid + off];
        __syncthreads();
    }

    if (tid == 0) {
        const int tok = inputs[b * SS + t];
        const float logp = lg[tok] - gmax - logf(ssum[0]);
        losses[t * BB + b] = -logp;
        sym[b * SS + t]    = (float)simx[0];
    }
}

// ---------------------------------------------------------------------------
// Deterministic fixed-order loss reduction: loss = sum(losses)/B
// ---------------------------------------------------------------------------
__global__ void k_finalize(const float* __restrict__ losses, float* __restrict__ out)
{
    if (threadIdx.x == 0 && blockIdx.x == 0) {
        float acc = 0.f;
        for (int i = 0; i < SS * BB; ++i) acc += losses[i];
        out[0] = acc / (float)BB;
    }
}

// ---------------------------------------------------------------------------
extern "C" void kernel_launch(void* const* d_in, const int* in_sizes, int n_in,
                              void* d_out, int out_size, void* d_ws, size_t ws_size,
                              hipStream_t stream)
{
    const int*   inputs  = (const int*)  d_in[0];
    const float* enc     = (const float*)d_in[1];
    const float* hidden0 = (const float*)d_in[2];
    const float* embt    = (const float*)d_in[3];
    const float* W_in    = (const float*)d_in[4];
    const float* b_in    = (const float*)d_in[5];
    const float* W_ih    = (const float*)d_in[6];
    const float* W_hh    = (const float*)d_in[7];
    const float* b_ih    = (const float*)d_in[8];
    const float* b_hh    = (const float*)d_in[9];
    const float* W_proj  = (const float*)d_in[10];
    const float* b_proj  = (const float*)d_in[11];
    const float* W_out   = (const float*)d_in[12];
    const float* b_out   = (const float*)d_in[13];

    char* ws = (char*)d_ws;
    size_t off = 0;
    auto take = [&](size_t bytes) -> void* {
        void* p = ws + off;
        off += (bytes + 255) & ~(size_t)255;
        return p;
    };

    // f16 swizzled weights (~42 MB; L2-resident on MI455X)
    _Float16* sw_Win   = (_Float16*)take((size_t)HH * (HH + EE) * 2);
    _Float16* sw_Wih   = (_Float16*)take((size_t)2 * 3 * HH * HH * 2);
    _Float16* sw_Whh   = (_Float16*)take((size_t)2 * 3 * HH * HH * 2);
    _Float16* sw_Wproj = (_Float16*)take((size_t)HH * (2 * HH + EE) * 2);
    _Float16* sw_Wout  = (_Float16*)take((size_t)VV * HH * 2);
    // f16 swizzled activation (A operand) scratch, double-buffered
    _Float16* aswA = (_Float16*)take((size_t)BB * (2 * HH + EE) * 2);
    _Float16* aswB = (_Float16*)take((size_t)BB * (2 * HH + EE) * 2);
    // fp32 activations / state
    float* h      = (float*)take((size_t)2 * BB * HH * 4);
    float* ctx    = (float*)take((size_t)BB * HH * 4);
    float* emb    = (float*)take((size_t)BB * EE * 4);
    float* xcat   = (float*)take((size_t)BB * (EE + HH) * 4);
    float* xbuf   = (float*)take((size_t)BB * HH * 4);
    float* gi     = (float*)take((size_t)BB * 3 * HH * 4);
    float* gh     = (float*)take((size_t)BB * 3 * HH * 4);
    float* pcat   = (float*)take((size_t)BB * (2 * HH + EE) * 4);
    float* proj   = (float*)take((size_t)BB * HH * 4);
    float* logits = (float*)take((size_t)BB * VV * 4);
    float* losses = (float*)take((size_t)SS * BB * 4);

    // One-time weight conversion + swizzle to WMMA operand layout.
    k_swizzle<<<dim3((HH + EE) / 32,   HH / 16),     32, 0, stream>>>(W_in,   sw_Win,   HH + EE);
    k_swizzle<<<dim3(HH / 32,          6 * HH / 16), 32, 0, stream>>>(W_ih,   sw_Wih,   HH);
    k_swizzle<<<dim3(HH / 32,          6 * HH / 16), 32, 0, stream>>>(W_hh,   sw_Whh,   HH);
    k_swizzle<<<dim3((2*HH + EE) / 32, HH / 16),     32, 0, stream>>>(W_proj, sw_Wproj, 2*HH + EE);
    k_swizzle<<<dim3(HH / 32,          VV / 16),     32, 0, stream>>>(W_out,  sw_Wout,  HH);

    // h <- hidden0
    hipMemcpyAsync(h, hidden0, (size_t)2 * BB * HH * 4, hipMemcpyDeviceToDevice, stream);

    // Layer-1 weight tile offset inside sw_Wih/sw_Whh: rows 1536..3071 ->
    // rt0 = 96 -> 96 * (512/32) tiles * 512 f16 each.
    const size_t l1off = (size_t)96 * 16 * 512;
    float* h0 = h;
    float* h1 = h + (size_t)BB * HH;

    float* sym = (float*)d_out + 1;

    for (int t = 0; t < SS; ++t) {
        // attention on top-layer hidden state h[1]
        k_attn<<<BB, 256, 0, stream>>>(h1, enc, ctx);
        k_embed_concat<<<BB, 256, 0, stream>>>(inputs, t, embt, ctx, xcat, emb);

        // x = [emb|ctx] @ W_in^T + b_in            (N=512, K=1024)
        k_swizzle<<<dim3((EE + HH) / 32, 2), 32, 0, stream>>>(xcat, aswA, EE + HH);
        k_wmma_gemm<<<HH / 128, 128, 0, stream>>>(aswA, sw_Win, b_in, xbuf, HH, EE + HH);

        // GRU layer 0: gi = x @ W_ih0^T, gh = h0 @ W_hh0^T   (N=1536, K=512)
        k_swizzle<<<dim3(HH / 32, 2), 32, 0, stream>>>(xbuf, aswA, HH);
        k_swizzle<<<dim3(HH / 32, 2), 32, 0, stream>>>(h0,   aswB, HH);
        k_wmma_gemm<<<3 * HH / 128, 128, 0, stream>>>(aswA, sw_Wih, b_ih, gi, 3 * HH, HH);
        k_wmma_gemm<<<3 * HH / 128, 128, 0, stream>>>(aswB, sw_Whh, b_hh, gh, 3 * HH, HH);
        k_gru_gates<<<BB * HH / 256, 256, 0, stream>>>(gi, gh, h0);

        // GRU layer 1 (input = freshly updated h0)
        k_swizzle<<<dim3(HH / 32, 2), 32, 0, stream>>>(h0, aswA, HH);
        k_swizzle<<<dim3(HH / 32, 2), 32, 0, stream>>>(h1, aswB, HH);
        k_wmma_gemm<<<3 * HH / 128, 128, 0, stream>>>(aswA, sw_Wih + l1off, b_ih + 3 * HH, gi, 3 * HH, HH);
        k_wmma_gemm<<<3 * HH / 128, 128, 0, stream>>>(aswB, sw_Whh + l1off, b_hh + 3 * HH, gh, 3 * HH, HH);
        k_gru_gates<<<BB * HH / 256, 256, 0, stream>>>(gi, gh, h1);

        // proj = [out|ctx|emb] @ W_proj^T + b_proj (N=512, K=1536)
        k_concat_proj<<<BB, 256, 0, stream>>>(h1, ctx, emb, pcat);
        k_swizzle<<<dim3((2 * HH + EE) / 32, 2), 32, 0, stream>>>(pcat, aswA, 2 * HH + EE);
        k_wmma_gemm<<<HH / 128, 128, 0, stream>>>(aswA, sw_Wproj, b_proj, proj, HH, 2 * HH + EE);

        // logits = proj @ W_out^T + b_out          (N=32000, K=512) — dominant GEMM
        k_swizzle<<<dim3(HH / 32, 2), 32, 0, stream>>>(proj, aswB, HH);
        k_wmma_gemm<<<VV / 128, 128, 0, stream>>>(aswB, sw_Wout, b_out, logits, VV, HH);

        k_lsm<<<BB, 256, 0, stream>>>(logits, inputs, t, losses, sym);
    }

    k_finalize<<<1, 32, 0, stream>>>(losses, (float*)d_out);
}
